// TransformerBlockClassical_65481071405559
// MI455X (gfx1250) — matrix operand order repaired and checked by reading, not compiled, not run on hardware
//
#include <hip/hip_runtime.h>
#include <math.h>

// ---------------------------------------------------------------------------
// CDNA5 (gfx1250) wave32 WMMA transformer block.
// All GEMMs run on v_wmma_f32_16x16x32_f16 (f16 operands, f32 accum).
// ---------------------------------------------------------------------------

typedef __attribute__((ext_vector_type(16))) _Float16 v16h;
typedef __attribute__((ext_vector_type(8)))  _Float16 v8h;
typedef __attribute__((ext_vector_type(4)))  _Float16 v4h;
typedef __attribute__((ext_vector_type(8)))  float    v8f;

#define WMMA_F16(a, b, c) \
  __builtin_amdgcn_wmma_f32_16x16x32_f16(false, (a), false, (b), (short)0, (c), false, false)

// Concatenate two 8-half (16B) chunks into one 16-half fragment register set.
static __device__ __forceinline__ v16h frag_cat(const _Float16* lo, const _Float16* hi) {
  union { v16h v; v8h p[2]; } u;
  u.p[0] = *(const v8h*)lo;
  u.p[1] = *(const v8h*)hi;
  return u.v;
}

// ---------------------------------------------------------------------------
// Generic WMMA GEMM: C[M,N] = epilogue(A[M,K] @ B[K,N] + bias)
//   MODE 0: C = acc                 (bias unused)
//   MODE 1: C = acc + bias
//   MODE 2: C = sigmoid(acc + bias)
//   MODE 3: C = (acc + bias) * aux[row,col]
// Block tile 128x128, 256 threads = 8 waves (4 in M, 2 in N),
// each wave owns a 32x64 sub-tile = 2x4 WMMA accumulators.
// ---------------------------------------------------------------------------
template <int MODE>
__global__ __launch_bounds__(256) void gemm_wmma(
    const float* __restrict__ A, const float* __restrict__ B,
    float* __restrict__ C, const float* __restrict__ bias,
    const float* __restrict__ aux, int M, int N, int K)
{
  __shared__ _Float16 As[128][40];   // [m][k], padded stride (80B) to spread banks
  __shared__ _Float16 Bs[128][40];   // B transposed: [n][k]

  const int t      = threadIdx.x;
  const int lane   = t & 31;
  const int wave   = t >> 5;
  const int wm     = wave >> 1;          // 0..3  (M direction)
  const int wn     = wave & 1;           // 0..1  (N direction)
  const int lrow   = lane & 15;
  const int hiHalf = (lane & 16) ? 1 : 0;
  const int kA     = hiHalf * 8;         // unused: A frags also load at kB .. kB+15
  const int kB     = hiHalf * 16;        // B-frag K offsets: kB .. kB+15

  const long m0 = (long)blockIdx.y * 128;
  const long n0 = (long)blockIdx.x * 128;

  v8f acc[2][4] = {};

  for (int k0 = 0; k0 < K; k0 += 32) {
    // Stage A tile 128x32 (f32 -> f16): 1024 float4 / 256 threads.
    #pragma unroll
    for (int i = 0; i < 4; ++i) {
      int idx = t + i * 256;
      int r = idx >> 3, c = (idx & 7) << 2;
      const float4 f = *(const float4*)(A + (m0 + r) * (long)K + k0 + c);
      v4h h4; h4[0] = (_Float16)f.x; h4[1] = (_Float16)f.y;
              h4[2] = (_Float16)f.z; h4[3] = (_Float16)f.w;
      *(v4h*)&As[r][c] = h4;
    }
    // Stage B tile 32x128 transposed into Bs[n][k].
    #pragma unroll
    for (int i = 0; i < 4; ++i) {
      int idx = t + i * 256;
      int kr = idx >> 5, c = (idx & 31) << 2;
      const float4 f = *(const float4*)(B + (long)(k0 + kr) * N + n0 + c);
      Bs[c + 0][kr] = (_Float16)f.x;
      Bs[c + 1][kr] = (_Float16)f.y;
      Bs[c + 2][kr] = (_Float16)f.z;
      Bs[c + 3][kr] = (_Float16)f.w;
    }
    // Pull the next k-tile toward the WGP while this one computes
    // (global_prefetch_b8; the 192MB L2 already holds the working set).
    if (k0 + 32 < K) {
      int r = t >> 3, c = (t & 7) << 2;
      __builtin_prefetch(A + (m0 + r) * (long)K + k0 + 32 + c, 0, 3);
      int kr = t >> 5, c2 = (t & 31) << 2;
      __builtin_prefetch(B + (long)(k0 + 32 + kr) * N + n0 + c2, 0, 3);
    }
    __syncthreads();

    v16h af[2], bf[4];
    #pragma unroll
    for (int m = 0; m < 2; ++m) {
      const _Float16* p = &As[wm * 32 + m * 16 + lrow][kB];
      af[m] = frag_cat(p, p + 8);
    }
    #pragma unroll
    for (int n = 0; n < 4; ++n) {
      const _Float16* p = &Bs[wn * 64 + n * 16 + lrow][kB];
      bf[n] = frag_cat(p, p + 8);
    }
    #pragma unroll
    for (int m = 0; m < 2; ++m)
      #pragma unroll
      for (int n = 0; n < 4; ++n)
        acc[m][n] = WMMA_F16(af[m], bf[n], acc[m][n]);

    __syncthreads();
  }

  // Epilogue. D layout: VGPR i -> row = i + hiHalf*8, col = lane%16.
  #pragma unroll
  for (int m = 0; m < 2; ++m) {
    #pragma unroll
    for (int n = 0; n < 4; ++n) {
      long col = n0 + wn * 64 + n * 16 + lrow;
      float bv = (MODE >= 1) ? bias[col] : 0.0f;
      long rbase = m0 + wm * 32 + m * 16 + hiHalf * 8;
      #pragma unroll
      for (int i = 0; i < 8; ++i) {
        long idx = (rbase + i) * (long)N + col;
        float v = acc[m][n][i] + bv;
        if (MODE == 2) v = 1.0f / (1.0f + __expf(-v));
        if (MODE == 3) v = v * aux[idx];
        C[idx] = v;
      }
    }
  }
}

// ---------------------------------------------------------------------------
// qkv (f32, [tok][3*1024]) -> q16/k16 [bh][s][64] f16, vT16 [bh][64][s] f16.
// ---------------------------------------------------------------------------
__global__ __launch_bounds__(256) void qkv_split(
    const float* __restrict__ qkv, _Float16* __restrict__ q16,
    _Float16* __restrict__ k16, _Float16* __restrict__ vT16)
{
  long gid = (long)blockIdx.x * 256 + threadIdx.x;  // B*S*H*HD = 4194304 exact
  int  d = gid & 63;
  int  h = (int)((gid >> 6) & 15);
  long s = (gid >> 10) & 2047;
  long b = gid >> 21;
  const float* base = qkv + (b * 2048 + s) * 3072 + h * 64 + d;
  long bh = b * 16 + h;
  q16 [(bh * 2048 + s) * 64 + d]   = (_Float16)base[0];
  k16 [(bh * 2048 + s) * 64 + d]   = (_Float16)base[1024];
  vT16[(bh * 64 + d) * 2048 + s]   = (_Float16)base[2048];
}

// ---------------------------------------------------------------------------
// Flash attention: one wave per 16-query tile, streaming 32 keys / iter.
// ---------------------------------------------------------------------------
__global__ __launch_bounds__(256) void attn_wmma(
    const _Float16* __restrict__ q16, const _Float16* __restrict__ k16,
    const _Float16* __restrict__ vT16, float* __restrict__ ctx)
{
  __shared__ _Float16 P[8][16][40];   // per-wave probability tile (16 rows x 32 keys)

  const int t      = threadIdx.x;
  const int lane   = t & 31;
  const int wave   = t >> 5;
  const int lrow   = lane & 15;
  const int hiHalf = (lane & 16) ? 1 : 0;
  const int kA     = hiHalf * 8;
  const int kB     = hiHalf * 16;
  const int rbase  = hiHalf * 8;

  int tile = blockIdx.x * 8 + wave;     // 4096 tiles = 32 (b,h) * 128 q-tiles
  int bh   = tile >> 7;
  int q0   = (tile & 127) << 4;

  const _Float16* qp = q16  + ((long)bh * 2048) * 64;
  const _Float16* kp = k16  + ((long)bh * 2048) * 64;
  const _Float16* vp = vT16 + ((long)bh * 64) * 2048;

  // Resident Q fragments (16x64 = two K-steps of 32).
  v16h qf[2];
  #pragma unroll
  for (int j = 0; j < 2; ++j) {
    const _Float16* p = qp + (long)(q0 + lrow) * 64 + j * 32 + kB;
    qf[j] = frag_cat(p, p + 8);
  }

  v8f   cacc[4] = {};
  float mrow[8], lsum[8];
  #pragma unroll
  for (int i = 0; i < 8; ++i) { mrow[i] = -1e30f; lsum[i] = 0.0f; }

  for (int kt = 0; kt < 2048; kt += 32) {
    // S = Q @ K^T for 32 keys: two 16x16 score tiles, inner dim 64.
    v8f s0 = {}, s1 = {};
    #pragma unroll
    for (int j = 0; j < 2; ++j) {
      const _Float16* p0 = kp + (long)(kt + lrow) * 64 + j * 32 + kB;
      s0 = WMMA_F16(qf[j], frag_cat(p0, p0 + 8), s0);
      const _Float16* p1 = kp + (long)(kt + 16 + lrow) * 64 + j * 32 + kB;
      s1 = WMMA_F16(qf[j], frag_cat(p1, p1 + 8), s1);
    }

    // Online softmax. Row r lives in accumulator slot i = r - rbase across
    // a 16-lane half, so stats are per-slot and reduced with xor-shuffles.
    #pragma unroll
    for (int i = 0; i < 8; ++i) {
      float a0 = s0[i] * 0.125f;          // 1/sqrt(64)
      float a1 = s1[i] * 0.125f;
      float rm = fmaxf(a0, a1);
      #pragma unroll
      for (int off = 1; off < 16; off <<= 1) rm = fmaxf(rm, __shfl_xor(rm, off, 32));
      float mn   = fmaxf(mrow[i], rm);
      float corr = __expf(mrow[i] - mn);
      mrow[i] = mn;
      float p0 = __expf(a0 - mn);
      float p1 = __expf(a1 - mn);
      float rs = p0 + p1;
      #pragma unroll
      for (int off = 1; off < 16; off <<= 1) rs += __shfl_xor(rs, off, 32);
      lsum[i] = lsum[i] * corr + rs;
      #pragma unroll
      for (int nn = 0; nn < 4; ++nn) cacc[nn][i] *= corr;
      // Scatter P (D-layout) into LDS row-major [qrow][key].
      P[wave][rbase + i][lrow]      = (_Float16)p0;
      P[wave][rbase + i][16 + lrow] = (_Float16)p1;
    }

    // Same-wave LDS RAW: make the scatter visible before fragment reload.
    asm volatile("s_wait_dscnt 0" ::: "memory");

    v16h pf;
    {
      const _Float16* p = &P[wave][lrow][kB];
      pf = frag_cat(p, p + 8);
    }

    // ctx += P @ V (V pre-transposed: column d is contiguous over keys).
    #pragma unroll
    for (int nn = 0; nn < 4; ++nn) {
      const _Float16* p = vp + (long)(nn * 16 + lrow) * 2048 + kt + kB;
      cacc[nn] = WMMA_F16(pf, frag_cat(p, p + 8), cacc[nn]);
    }
  }

  // Normalize and store ctx in token-major [b*2048+s][1024] layout.
  int b = bh >> 4, h = bh & 15;
  #pragma unroll
  for (int nn = 0; nn < 4; ++nn) {
    #pragma unroll
    for (int i = 0; i < 8; ++i) {
      long row = (long)b * 2048 + q0 + rbase + i;
      long col = h * 64 + nn * 16 + lrow;
      ctx[row * 1024 + col] = cacc[nn][i] / lsum[i];
    }
  }
}

// ---------------------------------------------------------------------------
// out[tok] = LayerNorm(x[tok] + y[tok]) * g + b     (E = 1024, 256 threads)
// ---------------------------------------------------------------------------
__global__ __launch_bounds__(256) void add_ln(
    const float* __restrict__ x, const float* __restrict__ y,
    const float* __restrict__ g, const float* __restrict__ be,
    float* __restrict__ out)
{
  __shared__ float red[8];
  const int  t    = threadIdx.x;
  const int  lane = t & 31;
  const int  wave = t >> 5;
  const long tok  = blockIdx.x;
  const float* xp = x + tok * 1024;
  const float* yp = y + tok * 1024;

  float v[4]; float s = 0.f;
  #pragma unroll
  for (int i = 0; i < 4; ++i) { v[i] = xp[t + i * 256] + yp[t + i * 256]; s += v[i]; }
  #pragma unroll
  for (int off = 1; off < 32; off <<= 1) s += __shfl_xor(s, off, 32);
  if (lane == 0) red[wave] = s;
  __syncthreads();
  s = red[0] + red[1] + red[2] + red[3] + red[4] + red[5] + red[6] + red[7];
  float mu = s * (1.0f / 1024.0f);

  float vs = 0.f;
  #pragma unroll
  for (int i = 0; i < 4; ++i) { float d = v[i] - mu; vs += d * d; }
  #pragma unroll
  for (int off = 1; off < 32; off <<= 1) vs += __shfl_xor(vs, off, 32);
  __syncthreads();
  if (lane == 0) red[wave] = vs;
  __syncthreads();
  vs = red[0] + red[1] + red[2] + red[3] + red[4] + red[5] + red[6] + red[7];
  float inv = rsqrtf(vs * (1.0f / 1024.0f) + 1e-5f);

  #pragma unroll
  for (int i = 0; i < 4; ++i) {
    int c = t + i * 256;
    out[tok * 1024 + c] = (v[i] - mu) * inv * g[c] + be[c];
  }
}

// ---------------------------------------------------------------------------
extern "C" void kernel_launch(void* const* d_in, const int* in_sizes, int n_in,
                              void* d_out, int out_size, void* d_ws, size_t ws_size,
                              hipStream_t stream)
{
  const float* x       = (const float*)d_in[0];
  const float* w_qkv   = (const float*)d_in[1];
  const float* w_o     = (const float*)d_in[2];
  const float* b_o     = (const float*)d_in[3];
  const float* w1      = (const float*)d_in[4];
  const float* b1      = (const float*)d_in[5];
  const float* w2      = (const float*)d_in[6];
  const float* b2      = (const float*)d_in[7];
  const float* w_ffn_o = (const float*)d_in[8];
  const float* b_ffn_o = (const float*)d_in[9];
  const float* g1      = (const float*)d_in[10];
  const float* be1     = (const float*)d_in[11];
  const float* g2      = (const float*)d_in[12];
  const float* be2     = (const float*)d_in[13];

  char* ws = (char*)d_ws;
  // Workspace plan (151 MB total, regions ping-ponged):
  //   [0,          50331648)  qkv          -> later gate (with ctx region)
  //   [50331648,   67108864)  ctx
  //   [67108864,   83886080)  x1
  //   [83886080,  150994944)  q16/k16/vT16 -> attn_out -> h
  float*    qkv      = (float*)(ws);
  float*    ctx      = (float*)(ws + 50331648);
  float*    x1       = (float*)(ws + 67108864);
  char*     regD     = ws + 83886080;
  _Float16* q16      = (_Float16*)(regD);
  _Float16* k16      = (_Float16*)(regD + 8388608);
  _Float16* vT16     = (_Float16*)(regD + 16777216);
  float*    attn_out = (float*)regD;       // reuses q16/k16 after attention
  float*    gbuf     = (float*)(ws);       // reuses qkv+ctx after LN1
  float*    hbuf     = (float*)regD;       // reuses attn_out region
  float*    ffn_out  = (float*)(ws);       // reuses gbuf after gated-h GEMM

  dim3 blk(256);

  // 1) qkv = x @ w_qkv                                   [4096x3072, K=1024]
  gemm_wmma<0><<<dim3(24, 32), blk, 0, stream>>>(x, w_qkv, qkv, nullptr, nullptr,
                                                 4096, 3072, 1024);
  // 2) split + f16 convert (v transposed)
  qkv_split<<<dim3(16384), blk, 0, stream>>>(qkv, q16, k16, vT16);
  // 3) attention -> ctx
  attn_wmma<<<dim3(512), blk, 0, stream>>>(q16, k16, vT16, ctx);
  // 4) attn_out = ctx @ w_o + b_o                        [4096x1024, K=1024]
  gemm_wmma<1><<<dim3(8, 32), blk, 0, stream>>>(ctx, w_o, attn_out, b_o, nullptr,
                                                4096, 1024, 1024);
  // 5) x1 = LN(x + attn_out)
  add_ln<<<dim3(4096), blk, 0, stream>>>(x, attn_out, g1, be1, x1);
  // 6) gate = sigmoid(x1 @ w2 + b2)                      [4096x4096, K=1024]
  gemm_wmma<2><<<dim3(32, 32), blk, 0, stream>>>(x1, w2, gbuf, b2, nullptr,
                                                 4096, 4096, 1024);
  // 7) h = (x1 @ w1 + b1) * gate
  gemm_wmma<3><<<dim3(32, 32), blk, 0, stream>>>(x1, w1, hbuf, b1, gbuf,
                                                 4096, 4096, 1024);
  // 8) ffn_out = h @ w_ffn_o + b_ffn_o                   [4096x1024, K=4096]
  gemm_wmma<1><<<dim3(8, 32), blk, 0, stream>>>(hbuf, w_ffn_o, ffn_out, b_ffn_o,
                                                nullptr, 4096, 1024, 4096);
  // 9) out = LN(x1 + ffn_out)
  add_ln<<<dim3(4096), blk, 0, stream>>>(x1, ffn_out, g2, be2, (float*)d_out);
}
